// ActorCriticGNN_RNN_Policy_61022895341568
// MI455X (gfx1250) — compile-verified, matrix operand order checked by reading, not run on hardware
//
#include <hip/hip_runtime.h>

// Problem constants (match reference)
#define AA   16        // agents / graphs
#define NN   1000      // nodes per graph
#define FF   16        // input feature dim
#define EE   16000     // edges per graph (before self loops)
#define NT   (AA*NN)   // 16000 total nodes
#define ETOT (AA*EE + NT) // 272000 batched edges incl. self loops
#define GD   128       // GNN / RNN dim
#define NACT 16

typedef __attribute__((ext_vector_type(16))) _Float16 v16h;
typedef __attribute__((ext_vector_type(8)))  float    v8f;

// ---- monotonic-uint float encoding for atomic max (global_atomic_max_u32) ----
__device__ __forceinline__ unsigned fenc(float x) {
  unsigned u = __float_as_uint(x);
  return (u & 0x80000000u) ? ~u : (u | 0x80000000u);
}
__device__ __forceinline__ float fdec(unsigned e) {
  unsigned u = (e & 0x80000000u) ? (e ^ 0x80000000u) : ~e;
  return __uint_as_float(u);
}

// batched edge t -> (src,dst) with per-graph node offsets + PyG self loops
__device__ __forceinline__ void edge_sd(int t, const int* __restrict__ ei, int& s, int& d) {
  if (t < AA*EE) {
    int a = t / EE, j = t - a*EE;
    s = ei[j]      + a*NN;   // edge_index[0][j]
    d = ei[EE + j] + a*NN;   // edge_index[1][j]
  } else {
    s = d = t - AA*EE;       // self loop
  }
}

__device__ __forceinline__ int agent_of(int d, const int* __restrict__ aidx) {
  int a = d / NN;
  return (aidx[a] == d) ? a : -1;
}

__device__ __forceinline__ float wred(float v) {
#pragma unroll
  for (int off = 16; off > 0; off >>= 1) v += __shfl_xor(v, off, 32);
  return v;
}

// ---------------- utility fills ----------------
__global__ void fill_u32(unsigned* __restrict__ p, unsigned v, int n) {
  int i = blockIdx.x * 256 + threadIdx.x;
  if (i < n) p[i] = v;
}

// ---------------- agent row discovery + needed-node mask ----------------
__global__ void find_agents(const float* __restrict__ nf, int* __restrict__ aidx) {
  int i = blockIdx.x * 256 + threadIdx.x;
  if (i >= NT) return;
  if (nf[i * FF] == 1.0f) aidx[i / NN] = i;   // exactly one per graph
}

__global__ void mark_needed(const int* __restrict__ ei, const int* __restrict__ aidx,
                            unsigned* __restrict__ needed) {
  int t = blockIdx.x * 256 + threadIdx.x;
  if (t >= ETOT) return;
  int s, d; edge_sd(t, ei, s, d);
  if (agent_of(d, aidx) >= 0) needed[s] = 1u;  // layer-1 output needed at s
}

// ---------------- weight packing into WMMA B-fragment order ----------------
// Fragment element j of lane L (col = L%16, hi = L/16) holds
//   k = (j<8) ? hi*8+j : 16 + hi*8 + (j-8)        (CDNA5 ISA 7.12.2, 16-bit B 32x16)
// Wp layout: Wp[(((nt*KT + kt)*32 + lane)*16) + j], f16, 32B-aligned per lane.
__global__ void pack_w(const float* __restrict__ W, _Float16* __restrict__ Wp,
                       int K, int Nc, int transB) {
  int KT = (K + 31) >> 5;
  int total = (Nc >> 4) * KT * 32 * 16;
  for (int i = blockIdx.x * 256 + threadIdx.x; i < total; i += gridDim.x * 256) {
    int j    = i & 15;
    int lane = (i >> 4) & 31;
    int kt   = (i >> 9) % KT;
    int nt   = (i >> 9) / KT;
    int col  = nt * 16 + (lane & 15);
    int hi   = lane >> 4;
    int k    = kt * 32 + ((j < 8) ? (hi * 8 + j) : (16 + hi * 8 + (j - 8)));
    float v = 0.f;
    if (k < K) v = transB ? W[col * K + k] : W[k * Nc + col];
    Wp[i] = (_Float16)v;
  }
}

// ---------------- WMMA GEMM: Y[M,Nc] = X[M,K] @ W(packed) ----------------
// grid.x = M/16, block = 256 (8 waves).  K is compile-time (16 or 128) so the
// K-loop fully unrolls with no guards: af = one 32B LDS vector load from the
// fragment-order swizzled A tile, bf = one 32B global vector load from Wp.
template<int K>
__global__ void gemm_wmma(const float* __restrict__ X, const _Float16* __restrict__ Wp,
                          float* __restrict__ Y, int Nc) {
  constexpr int KT = (K + 31) / 32;
  __shared__ alignas(32) _Float16 Asw[KT][32][16];
  const int tid = threadIdx.x, mtile = blockIdx.x;

  if (K % 32 != 0) {            // zero-pad tail K elements (K=16 case)
    for (int idx = tid; idx < KT * 32 * 16; idx += blockDim.x)
      (&Asw[0][0][0])[idx] = (_Float16)0.f;
    __syncthreads();
  }
  // coalesced global read of the 16xK A tile, scatter into fragment order:
  // kk = k%32 -> hi = (kk>>3)&1, j = (kk&7) + (kk>=16 ? 8 : 0), lane = r + 16*hi
  for (int idx = tid; idx < 16 * K; idx += blockDim.x) {
    int r = idx / K, k = idx - r * K;
    int kt = k >> 5, kk = k & 31;
    int hi = (kk >> 3) & 1;
    int j  = (kk & 7) + ((kk >> 4) << 3);
    Asw[kt][r + 16 * hi][j] = (_Float16)X[(mtile * 16 + r) * K + k];
  }
  __syncthreads();

  const int lane = tid & 31, wave = tid >> 5, nwaves = blockDim.x >> 5;
  const int row = lane & 15, hi = lane >> 4;
  const int ntiles = Nc >> 4;

  for (int nt = wave; nt < ntiles; nt += nwaves) {
    v8f c = {};
#pragma unroll
    for (int kt = 0; kt < KT; ++kt) {
      v16h af = *(const v16h*)(&Asw[kt][lane][0]);
      v16h bf = *(const v16h*)(&Wp[(((size_t)nt * KT + kt) * 32 + lane) * 16]);
      c = __builtin_amdgcn_wmma_f32_16x16x32_f16(false, af, false, bf,
                                                 (short)0, c, false, false);
    }
#pragma unroll
    for (int r = 0; r < 8; ++r) {
      int gr = mtile * 16 + r + 8 * hi;       // C/D layout: VGPR r -> row r+8*hi
      Y[gr * Nc + nt * 16 + row] = c[r];
    }
  }
}

// ---------------- Layer-1 GATv2 edge passes (4 heads x 32 ch), dst in needed set ----------------
__global__ void e1_logits(const float* __restrict__ xl, const float* __restrict__ xr,
                          const float* __restrict__ att, const int* __restrict__ ei,
                          const unsigned* __restrict__ needed,
                          float* __restrict__ e1, unsigned* __restrict__ m1) {
  int gt = blockIdx.x * 256 + threadIdx.x;
  int t = gt >> 5, lane = gt & 31;
  if (t >= ETOT) return;
  int s, d; edge_sd(t, ei, s, d);
  if (!needed[d]) return;
  float p[4];
#pragma unroll
  for (int q = 0; q < 4; ++q) {
    int ch = q * 32 + lane;
    float v = xl[s * GD + ch] + xr[d * GD + ch];
    v = (v > 0.f) ? v : 0.2f * v;              // leaky relu, slope 0.2
    p[q] = v * att[ch];                        // att1 is [4,32] flat
  }
#pragma unroll
  for (int q = 0; q < 4; ++q) p[q] = wred(p[q]);
  if (lane < 4) {
    float ev = (lane == 0) ? p[0] : (lane == 1) ? p[1] : (lane == 2) ? p[2] : p[3];
    e1[t * 4 + lane] = ev;
    atomicMax(&m1[d * 4 + lane], fenc(ev));
  }
}

__global__ void e1_soft(const int* __restrict__ ei, const unsigned* __restrict__ needed,
                        const unsigned* __restrict__ m1,
                        float* __restrict__ e1, float* __restrict__ s1) {
  int gt = blockIdx.x * 256 + threadIdx.x;
  if (gt >= ETOT * 4) return;
  int t = gt >> 2, h = gt & 3;
  int s, d; edge_sd(t, ei, s, d); (void)s;
  if (!needed[d]) return;
  float av = __expf(e1[gt] - fdec(m1[d * 4 + h]));
  e1[gt] = av;                                  // reuse buffer for numerator
  atomicAdd(&s1[d * 4 + h], av);
}

__global__ void e1_aggr(const float* __restrict__ xl, const int* __restrict__ ei,
                        const unsigned* __restrict__ needed,
                        const float* __restrict__ e1, const float* __restrict__ s1,
                        float* __restrict__ out1) {
  int gt = blockIdx.x * 256 + threadIdx.x;
  int t = gt >> 5, lane = gt & 31;
  if (t >= ETOT) return;
  int s, d; edge_sd(t, ei, s, d);
  if (!needed[d]) return;
#pragma unroll
  for (int q = 0; q < 4; ++q) {
    float alpha = e1[t * 4 + q] / s1[d * 4 + q];
    int ch = q * 32 + lane;
    atomicAdd(&out1[d * GD + ch], alpha * xl[s * GD + ch]);
  }
}

// ---------------- LayerNorm + ReLU (wave per node) ----------------
__global__ void ln_relu(const float* __restrict__ out1, const float* __restrict__ b1,
                        const float* __restrict__ lw, const float* __restrict__ lb,
                        float* __restrict__ hln) {
  int gt = blockIdx.x * 256 + threadIdx.x;
  int n = gt >> 5, lane = gt & 31;
  if (n >= NT) return;
  float v[4];
#pragma unroll
  for (int q = 0; q < 4; ++q) v[q] = out1[n * GD + q * 32 + lane] + b1[q * 32 + lane];
  float mu = wred(v[0] + v[1] + v[2] + v[3]) * (1.f / GD);
  float var = 0.f;
#pragma unroll
  for (int q = 0; q < 4; ++q) { float dd = v[q] - mu; var += dd * dd; }
  var = wred(var) * (1.f / GD);
  float inv = rsqrtf(var + 1e-5f);
#pragma unroll
  for (int q = 0; q < 4; ++q) {
    int ch = q * 32 + lane;
    float o = (v[q] - mu) * inv * lw[ch] + lb[ch];
    hln[n * GD + ch] = (o > 0.f) ? o : 0.f;
  }
}

// ---------------- Layer-2 GATv2 edge passes (1 head x 128 ch), dst must be an agent node ----------------
__global__ void e2_logits(const float* __restrict__ xl, const float* __restrict__ xr,
                          const float* __restrict__ att, const int* __restrict__ ei,
                          const int* __restrict__ aidx,
                          float* __restrict__ e2, unsigned* __restrict__ m2) {
  int gt = blockIdx.x * 256 + threadIdx.x;
  int t = gt >> 5, lane = gt & 31;
  if (t >= ETOT) return;
  int s, d; edge_sd(t, ei, s, d);
  int a = agent_of(d, aidx);
  if (a < 0) return;
  float p = 0.f;
#pragma unroll
  for (int q = 0; q < 4; ++q) {
    int ch = q * 32 + lane;
    float v = xl[s * GD + ch] + xr[d * GD + ch];
    v = (v > 0.f) ? v : 0.2f * v;
    p += v * att[ch];
  }
  p = wred(p);
  if (lane == 0) { e2[t] = p; atomicMax(&m2[a], fenc(p)); }
}

__global__ void e2_soft(const int* __restrict__ ei, const int* __restrict__ aidx,
                        const unsigned* __restrict__ m2,
                        float* __restrict__ e2, float* __restrict__ s2) {
  int t = blockIdx.x * 256 + threadIdx.x;
  if (t >= ETOT) return;
  int s, d; edge_sd(t, ei, s, d); (void)s;
  int a = agent_of(d, aidx);
  if (a < 0) return;
  float av = __expf(e2[t] - fdec(m2[a]));
  e2[t] = av;
  atomicAdd(&s2[a], av);
}

__global__ void e2_aggr(const float* __restrict__ xl, const int* __restrict__ ei,
                        const int* __restrict__ aidx,
                        const float* __restrict__ e2, const float* __restrict__ s2,
                        float* __restrict__ afeat) {
  int gt = blockIdx.x * 256 + threadIdx.x;
  int t = gt >> 5, lane = gt & 31;
  if (t >= ETOT) return;
  int s, d; edge_sd(t, ei, s, d);
  int a = agent_of(d, aidx);
  if (a < 0) return;
  float alpha = e2[t] / s2[a];
#pragma unroll
  for (int q = 0; q < 4; ++q) {
    int ch = q * 32 + lane;
    atomicAdd(&afeat[a * GD + ch], alpha * xl[s * GD + ch]);
  }
}

__global__ void add_b2(float* __restrict__ afeat, const float* __restrict__ b2) {
  int i = blockIdx.x * 256 + threadIdx.x;
  if (i < AA * GD) afeat[i] += b2[i & 127];
}

// ---------------- GRU elementwise + heads ----------------
__global__ void gru_elem(const float* __restrict__ gi, const float* __restrict__ gh,
                         const float* __restrict__ bih, const float* __restrict__ bhh,
                         const float* __restrict__ h0, float* __restrict__ nh) {
  int i = blockIdx.x * 256 + threadIdx.x;
  if (i >= AA * GD) return;
  int a = i >> 7, dd = i & 127;
  float ir  = gi[a * 384 + dd]       + bih[dd];
  float iz  = gi[a * 384 + 128 + dd] + bih[128 + dd];
  float in_ = gi[a * 384 + 256 + dd] + bih[256 + dd];
  float hr  = gh[a * 384 + dd]       + bhh[dd];
  float hz  = gh[a * 384 + 128 + dd] + bhh[128 + dd];
  float hn  = gh[a * 384 + 256 + dd] + bhh[256 + dd];
  float r = 1.f / (1.f + __expf(-(ir + hr)));
  float z = 1.f / (1.f + __expf(-(iz + hz)));
  float n = tanhf(in_ + r * hn);
  nh[i] = (1.f - z) * n + z * h0[i];
}

__global__ void heads(const float* __restrict__ nh, const float* __restrict__ Wa,
                      const float* __restrict__ ba, const float* __restrict__ Wv,
                      const float* __restrict__ bv, const int* __restrict__ amask,
                      float* __restrict__ out) {
  int tid = threadIdx.x;
  if (tid < AA * NACT) {
    int a = tid >> 4, j = tid & 15;
    float acc = 0.f;
    for (int k = 0; k < GD; ++k) acc += nh[a * GD + k] * Wa[k * NACT + j];
    acc += ba[j];
    if (amask[a * NACT + j] == 0) acc = -1e8f;
    out[a * NACT + j] = acc;
  }
  if (tid < AA) {
    float acc = 0.f;
    for (int k = 0; k < GD; ++k) acc += nh[tid * GD + k] * Wv[k];
    out[AA * NACT + tid] = acc + bv[0];
  }
}

// ---------------- host launcher ----------------
extern "C" void kernel_launch(void* const* d_in, const int* in_sizes, int n_in,
                              void* d_out, int out_size, void* d_ws, size_t ws_size,
                              hipStream_t stream) {
  (void)in_sizes; (void)n_in; (void)out_size;
  const float* nf    = (const float*)d_in[0];
  const int*   ei    = (const int*)  d_in[1];
  const float* h0    = (const float*)d_in[2];
  const int*   amask = (const int*)  d_in[3];
  const float* Wl1   = (const float*)d_in[4];
  const float* Wr1   = (const float*)d_in[5];
  const float* att1  = (const float*)d_in[6];
  const float* b1    = (const float*)d_in[7];
  const float* lnw   = (const float*)d_in[8];
  const float* lnb   = (const float*)d_in[9];
  const float* Wl2   = (const float*)d_in[10];
  const float* Wr2   = (const float*)d_in[11];
  const float* att2  = (const float*)d_in[12];
  const float* b2    = (const float*)d_in[13];
  const float* W_ih  = (const float*)d_in[14];
  const float* W_hh  = (const float*)d_in[15];
  const float* b_ih  = (const float*)d_in[16];
  const float* b_hh  = (const float*)d_in[17];
  const float* Wa    = (const float*)d_in[18];
  const float* ba    = (const float*)d_in[19];
  const float* Wv    = (const float*)d_in[20];
  const float* bv    = (const float*)d_in[21];
  float* out = (float*)d_out;

  // workspace carve-up (256B-aligned chunks)
  char* wsp = (char*)d_ws;
  auto alloc = [&](size_t bytes) {
    void* p = (void*)wsp;
    wsp += (bytes + 255) & ~(size_t)255;
    return p;
  };
  float*    xl1    = (float*)alloc((size_t)NT * GD * 4);
  float*    xr1    = (float*)alloc((size_t)NT * GD * 4);
  float*    out1   = (float*)alloc((size_t)NT * GD * 4);
  float*    hln    = (float*)alloc((size_t)NT * GD * 4);
  float*    xl2    = (float*)alloc((size_t)NT * GD * 4);
  float*    xr2    = (float*)alloc((size_t)NT * GD * 4);
  float*    e1     = (float*)alloc((size_t)ETOT * 4 * 4);
  float*    s1     = (float*)alloc((size_t)NT * 4 * 4);
  unsigned* m1     = (unsigned*)alloc((size_t)NT * 4 * 4);
  float*    e2     = (float*)alloc((size_t)ETOT * 4);
  float*    s2     = (float*)alloc(AA * 4);
  unsigned* m2     = (unsigned*)alloc(AA * 4);
  unsigned* needed = (unsigned*)alloc(NT * 4);
  int*      aidx   = (int*)alloc(AA * 4);
  float*    afeat  = (float*)alloc((size_t)AA * GD * 4);
  float*    gi     = (float*)alloc((size_t)AA * 384 * 4);
  float*    gh     = (float*)alloc((size_t)AA * 384 * 4);
  // packed f16 weights (fragment order)
  const int PK1 = 8 * 1 * 32 * 16;      // K=16  -> KT=1, Nc=128
  const int PK2 = 8 * 4 * 32 * 16;      // K=128 -> KT=4, Nc=128
  const int PKG = 24 * 4 * 32 * 16;     // K=128 -> KT=4, Nc=384
  _Float16* Wp_l1 = (_Float16*)alloc((size_t)PK1 * 2);
  _Float16* Wp_r1 = (_Float16*)alloc((size_t)PK1 * 2);
  _Float16* Wp_l2 = (_Float16*)alloc((size_t)PK2 * 2);
  _Float16* Wp_r2 = (_Float16*)alloc((size_t)PK2 * 2);
  _Float16* Wp_ih = (_Float16*)alloc((size_t)PKG * 2);
  _Float16* Wp_hh = (_Float16*)alloc((size_t)PKG * 2);
  if ((size_t)(wsp - (char*)d_ws) > ws_size) return;  // insufficient scratch

  auto blk = [](long long n) { return dim3((unsigned)((n + 255) / 256)); };

  // re-init buffers that must start at zero each call
  fill_u32<<<blk(NT * GD), 256, 0, stream>>>((unsigned*)out1, 0u, NT * GD);
  fill_u32<<<blk(NT * 4), 256, 0, stream>>>((unsigned*)s1, 0u, NT * 4);
  fill_u32<<<blk(NT * 4), 256, 0, stream>>>(m1, 0u, NT * 4);
  fill_u32<<<blk(NT), 256, 0, stream>>>(needed, 0u, NT);
  fill_u32<<<blk(AA), 256, 0, stream>>>((unsigned*)s2, 0u, AA);
  fill_u32<<<blk(AA), 256, 0, stream>>>(m2, 0u, AA);
  fill_u32<<<blk(AA * GD), 256, 0, stream>>>((unsigned*)afeat, 0u, AA * GD);

  find_agents<<<blk(NT), 256, 0, stream>>>(nf, aidx);
  mark_needed<<<blk(ETOT), 256, 0, stream>>>(ei, aidx, needed);

  // pack weights into WMMA fragment order (f16)
  pack_w<<<blk(PK1), 256, 0, stream>>>(Wl1,  Wp_l1, FF, GD,      0);
  pack_w<<<blk(PK1), 256, 0, stream>>>(Wr1,  Wp_r1, FF, GD,      0);
  pack_w<<<blk(PK2), 256, 0, stream>>>(Wl2,  Wp_l2, GD, GD,      0);
  pack_w<<<blk(PK2), 256, 0, stream>>>(Wr2,  Wp_r2, GD, GD,      0);
  pack_w<<<blk(PKG), 256, 0, stream>>>(W_ih, Wp_ih, GD, 3 * GD,  1);
  pack_w<<<blk(PKG), 256, 0, stream>>>(W_hh, Wp_hh, GD, 3 * GD,  1);

  // layer-1 feature transforms (WMMA), x is [NT,16]
  gemm_wmma<FF><<<NT / 16, 256, 0, stream>>>(nf, Wp_l1, xl1, GD);
  gemm_wmma<FF><<<NT / 16, 256, 0, stream>>>(nf, Wp_r1, xr1, GD);

  // layer-1 attention restricted to needed dst nodes
  e1_logits<<<blk((long long)ETOT * 32), 256, 0, stream>>>(xl1, xr1, att1, ei, needed, e1, m1);
  e1_soft<<<blk((long long)ETOT * 4), 256, 0, stream>>>(ei, needed, m1, e1, s1);
  e1_aggr<<<blk((long long)ETOT * 32), 256, 0, stream>>>(xl1, ei, needed, e1, s1, out1);

  ln_relu<<<blk((long long)NT * 32), 256, 0, stream>>>(out1, b1, lnw, lnb, hln);

  // layer-2 transforms (WMMA)
  gemm_wmma<GD><<<NT / 16, 256, 0, stream>>>(hln, Wp_l2, xl2, GD);
  gemm_wmma<GD><<<NT / 16, 256, 0, stream>>>(hln, Wp_r2, xr2, GD);

  // layer-2 attention restricted to agent dst nodes; aggregates directly into afeat
  e2_logits<<<blk((long long)ETOT * 32), 256, 0, stream>>>(xl2, xr2, att2, ei, aidx, e2, m2);
  e2_soft<<<blk(ETOT), 256, 0, stream>>>(ei, aidx, m2, e2, s2);
  e2_aggr<<<blk((long long)ETOT * 32), 256, 0, stream>>>(xl2, ei, aidx, e2, s2, afeat);

  add_b2<<<blk(AA * GD), 256, 0, stream>>>(afeat, b2);

  // GRU matmuls via WMMA: gi = afeat @ W_ih^T, gh = h0 @ W_hh^T   (M=16,K=128,N=384)
  gemm_wmma<GD><<<1, 256, 0, stream>>>(afeat, Wp_ih, gi, 3 * GD);
  gemm_wmma<GD><<<1, 256, 0, stream>>>(h0,    Wp_hh, gh, 3 * GD);

  // next_h lives at out+272 (after 256 logits + 16 values)
  float* nh = out + AA * NACT + AA;
  gru_elem<<<blk(AA * GD), 256, 0, stream>>>(gi, gh, b_ih, b_hh, h0, nh);
  heads<<<1, 256, 0, stream>>>(nh, Wa, ba, Wv, bv, amask, out);
}
